// CongestioNN_4982162063702
// MI455X (gfx1250) — compile-verified, matrix operand order checked by reading, not run on hardware
//
#include <hip/hip_runtime.h>
#include <hip/hip_bf16.h>

// ---------------------------------------------------------------------------
// Problem constants (from the reference)
// ---------------------------------------------------------------------------
constexpr int N_ = 100000;
constexpr int E_ = 1600000;
constexpr int D_ = 32;
constexpr int L_ = 2;
constexpr float EPS_ = 1e-5f;

constexpr int WPB = 8;                 // waves per block (256 threads, wave32)
constexpr int ETILES = E_ / 16;        // 100000 (exact)
constexpr int NTILES = N_ / 16;        // 6250  (exact)
constexpr int EDGE_BLOCKS = ETILES / WPB;        // 12500 (exact -> __syncthreads safe)
constexpr int NODE_BLOCKS = (NTILES + WPB - 1) / WPB; // 782 (has idle waves -> no block barrier)

// ---------------------------------------------------------------------------
// WMMA types / helpers (CDNA5 gfx1250, wave32)
// ---------------------------------------------------------------------------
typedef __attribute__((ext_vector_type(16))) _Float16 v16h;
typedef __attribute__((ext_vector_type(8)))  _Float16 v8h;
typedef __attribute__((ext_vector_type(8)))  float    v8f;

__device__ __forceinline__ v8f wmma_f16(v16h a, v16h b, v8f c) {
    // D = A(16x32 f16) x B(32x16 f16) + C(16x16 f32)
    return __builtin_amdgcn_wmma_f32_16x16x32_f16(
        false, a, false, b, (short)0, c, false, false);
}

// swish via hardware transcendentals: v_exp_f32 + v_rcp_f32 + v_mul_f32.
// Avoids the IEEE div_scale/div_fmas expansion of a full float division.
__device__ __forceinline__ float swishf(float x) {
    float e = __expf(-x);
    return x * __builtin_amdgcn_rcpf(1.0f + e);
}

// A-fragment (16x32, 16-bit) lane layout per ISA 7.12.2:
//   lane l: row m = l&15, half = l>>4
//   elems 0..7  -> K = half*8 + (0..7)
//   elems 8..15 -> K = 16 + half*8 + (0..7)
__device__ __forceinline__ v16h load_rowfrag(const _Float16* __restrict__ row, int half) {
    v8h lo = *(const v8h*)(row + half * 8);
    v8h hi = *(const v8h*)(row + 16 + half * 8);
    v16h a;
#pragma unroll
    for (int i = 0; i < 8; ++i) { a[i] = lo[i]; a[8 + i] = hi[i]; }
    return a;
}

// Wave-local LDS sync: DS ops from the same wave are processed in order, so a
// full dscnt drain makes prior lane-crossing LDS stores visible to loads.
__device__ __forceinline__ void wave_lds_sync() {
    __builtin_amdgcn_wave_barrier();
    asm volatile("s_wait_dscnt 0" ::: "memory");
    __builtin_amdgcn_wave_barrier();
}

__device__ __forceinline__ void atomic_add_f32(float* p, float v) {
    __hip_atomic_fetch_add(p, v, __ATOMIC_RELAXED, __HIP_MEMORY_SCOPE_AGENT);
}

// ---------------------------------------------------------------------------
// Workspace layout
// ---------------------------------------------------------------------------
constexpr size_t alignup(size_t x) { return (x + 255) & ~size_t(255); }
constexpr size_t SZ_WFRAG = size_t(L_) * 12 * 512 * 2;              // 24 KB f16 B-fragments
constexpr size_t OFF_H    = alignup(SZ_WFRAG);                      // h   f32 [N,32]
constexpr size_t OFF_H16  = alignup(OFF_H   + size_t(N_) * D_ * 4); // h16 f16 [N,32]
constexpr size_t OFF_AGG  = alignup(OFF_H16 + size_t(N_) * D_ * 2); // agg f32 [N,32]
constexpr size_t OFF_DEG  = alignup(OFF_AGG + size_t(N_) * D_ * 4); // recip deg f32 [N]
constexpr size_t OFF_BNS  = alignup(OFF_DEG + size_t(N_) * 4);      // BN sums [64]
constexpr size_t OFF_BNP  = alignup(OFF_BNS + 64 * 4);              // BN scale/shift [64]
constexpr size_t OFF_T16  = alignup(OFF_BNP + 64 * 4);              // t16 f16 [ETILES][512]

// ---------------------------------------------------------------------------
// Weight pre-swizzle: B-fragment layout (mirror of A, per ISA sparse-B tables):
//   lane l: col n = nt*16 + (l&15), half = l>>4
//   elem i: k = kc*32 + 16*(i>>3) + half*8 + (i&7)
// Per layer 12 fragments of 512 halfs: msg_w1[4], msg_w2[2], upd_w1[4], upd_w2[2]
// ---------------------------------------------------------------------------
__global__ void __launch_bounds__(256) prep_weights_kernel(
    const float* __restrict__ msg_w1, const float* __restrict__ msg_w2,
    const float* __restrict__ upd_w1, const float* __restrict__ upd_w2,
    _Float16* __restrict__ wfrag)
{
    int gid = blockIdx.x * 256 + threadIdx.x;          // 24*512 = 12288 threads
    if (gid >= L_ * 12 * 512) return;
    int frag = gid >> 9;
    int lane = (gid >> 4) & 31;
    int i    = gid & 15;
    int layer = frag / 12;
    int r     = frag % 12;
    const float* W; int kc, nt;
    if (r < 4)       { W = msg_w1 + size_t(layer) * 64 * 32; kc = r >> 1;       nt = r & 1; }
    else if (r < 6)  { W = msg_w2 + size_t(layer) * 32 * 32; kc = 0;            nt = r - 4; }
    else if (r < 10) { int q = r - 6;
                       W = upd_w1 + size_t(layer) * 64 * 32; kc = q >> 1;       nt = q & 1; }
    else             { W = upd_w2 + size_t(layer) * 32 * 32; kc = 0;            nt = r - 10; }
    int half = lane >> 4;
    int n = nt * 16 + (lane & 15);
    int k = kc * 32 + 16 * (i >> 3) + half * 8 + (i & 7);
    wfrag[size_t(frag) * 512 + lane * 16 + i] = (_Float16)W[k * 32 + n];
}

// ---------------------------------------------------------------------------
// Embedding: h = x @ emb_w + emb_b (N x 4 @ 4 x 32); also f16 mirror
// ---------------------------------------------------------------------------
__global__ void __launch_bounds__(256) embed_kernel(
    const float* __restrict__ x, const float* __restrict__ w, const float* __restrict__ b,
    float* __restrict__ h, _Float16* __restrict__ h16)
{
    int gid = blockIdx.x * 256 + threadIdx.x;
    if (gid >= N_ * D_) return;
    int i = gid >> 5, j = gid & 31;
    const float* xr = x + size_t(i) * 4;
    float acc = b[j];
#pragma unroll
    for (int k = 0; k < 4; ++k) acc = fmaf(xr[k], w[k * 32 + j], acc);
    h[gid] = acc;
    h16[gid] = (_Float16)acc;
}

__global__ void __launch_bounds__(256) fill_kernel(float* __restrict__ p, int n) {
    int g = blockIdx.x * blockDim.x + threadIdx.x;
    if (g < n) p[g] = 0.0f;
}

__global__ void __launch_bounds__(256) degree_kernel(const int* __restrict__ dst,
                                                     float* __restrict__ cnt) {
    int e = blockIdx.x * 256 + threadIdx.x;
    if (e >= E_) return;
    atomic_add_f32(&cnt[dst[e]], 1.0f);
}

__global__ void __launch_bounds__(256) recip_kernel(float* __restrict__ cnt) {
    int i = blockIdx.x * 256 + threadIdx.x;
    if (i >= N_) return;
    cnt[i] = __builtin_amdgcn_rcpf(fmaxf(cnt[i], 1.0f));
}

// ---------------------------------------------------------------------------
// Edge MLP stage 1: t = swish(cat(h[dst],h[src]) @ W1 + b1)
// One 16-edge tile per wave. Stores t16 in A-fragment layout (contiguous 32B
// per lane) + accumulates BN sum / sumsq via LDS then global f32 atomics.
// Grid is exact (all waves active) -> __syncthreads is safe.
// ---------------------------------------------------------------------------
__global__ void __launch_bounds__(256) edge_msg1_kernel(
    const _Float16* __restrict__ h16,
    const int* __restrict__ src, const int* __restrict__ dst,
    const _Float16* __restrict__ wf1, const float* __restrict__ b1,
    _Float16* __restrict__ t16, float* __restrict__ bnsum /*[64]*/)
{
    __shared__ _Float16 tb[WPB][16][32];
    __shared__ float bsum[32], bsq[32];
    int tid = threadIdx.x;
    if (tid < 32) { bsum[tid] = 0.0f; bsq[tid] = 0.0f; }
    __syncthreads();

    int lane = tid & 31, wv = tid >> 5;
    int tile = blockIdx.x * WPB + wv;
    int m = lane & 15, half = lane >> 4;
    int e = tile * 16 + m;
    int di = dst[e], si = src[e];

    v16h a0 = load_rowfrag(h16 + size_t(di) * 32, half);   // x_i = h[dst], K 0..31
    v16h a1 = load_rowfrag(h16 + size_t(si) * 32, half);   // x_j = h[src], K 32..63
    v16h b00 = *(const v16h*)(wf1 + 0 * 512 + lane * 16);  // kc0 nt0
    v16h b01 = *(const v16h*)(wf1 + 1 * 512 + lane * 16);  // kc0 nt1
    v16h b10 = *(const v16h*)(wf1 + 2 * 512 + lane * 16);  // kc1 nt0
    v16h b11 = *(const v16h*)(wf1 + 3 * 512 + lane * 16);  // kc1 nt1

    v8f c0 = {}, c1 = {};
    c0 = wmma_f16(a0, b00, c0); c0 = wmma_f16(a1, b10, c0);
    c1 = wmma_f16(a0, b01, c1); c1 = wmma_f16(a1, b11, c1);

    int n0 = lane & 15;
    float bias0 = b1[n0], bias1 = b1[16 + n0];
    float s0 = 0.f, q0 = 0.f, s1 = 0.f, q1 = 0.f;
#pragma unroll
    for (int r = 0; r < 8; ++r) {
        int mr = (lane >> 4) * 8 + r;                      // C layout: lanes16-31 -> M=8..15
        float t0 = swishf(c0[r] + bias0);
        float t1 = swishf(c1[r] + bias1);
        tb[wv][mr][n0]      = (_Float16)t0;
        tb[wv][mr][16 + n0] = (_Float16)t1;
        s0 += t0; q0 += t0 * t0; s1 += t1; q1 += t1 * t1;
    }
    atomicAdd(&bsum[n0], s0);      atomicAdd(&bsq[n0], q0);
    atomicAdd(&bsum[16 + n0], s1); atomicAdd(&bsq[16 + n0], q1);
    __syncthreads();

    // Re-read in A-fragment order, store 32 contiguous bytes per lane.
    v16h out;
#pragma unroll
    for (int i = 0; i < 8; ++i) {
        out[i]     = tb[wv][m][half * 8 + i];
        out[8 + i] = tb[wv][m][16 + half * 8 + i];
    }
    *(v16h*)(t16 + size_t(tile) * 512 + lane * 16) = out;

    if (tid < 32) {
        atomic_add_f32(&bnsum[tid],      bsum[tid]);
        atomic_add_f32(&bnsum[32 + tid], bsq[tid]);
    }
}

__global__ void bn_finalize_kernel(const float* __restrict__ bnsum,
                                   const float* __restrict__ g, const float* __restrict__ b,
                                   float* __restrict__ ss /*[64] scale,shift*/)
{
    int n = threadIdx.x;
    if (n >= 32) return;
    float mu  = bnsum[n] * (1.0f / E_);
    float var = bnsum[32 + n] * (1.0f / E_) - mu * mu;    // biased var (torch BN training)
    float sc  = g[n] * rsqrtf(var + EPS_);
    ss[n] = sc;
    ss[32 + n] = b[n] - mu * sc;
}

// ---------------------------------------------------------------------------
// Edge MLP stage 2: msg = swish(BN(t) @ W2 + b2); scatter-sum into agg[dst]
// ---------------------------------------------------------------------------
__global__ void __launch_bounds__(256) edge_msg2_kernel(
    const _Float16* __restrict__ t16, const int* __restrict__ dst,
    const float* __restrict__ ss, const _Float16* __restrict__ wf2,
    const float* __restrict__ b2, float* __restrict__ agg)
{
    __shared__ float sc[32], sh[32];
    int tid = threadIdx.x;
    if (tid < 32) { sc[tid] = ss[tid]; sh[tid] = ss[32 + tid]; }
    __syncthreads();

    int lane = tid & 31, wv = tid >> 5;
    int tile = blockIdx.x * WPB + wv;
    int half = lane >> 4;

    v16h a = *(const v16h*)(t16 + size_t(tile) * 512 + lane * 16);
    v16h an;
#pragma unroll
    for (int i = 0; i < 16; ++i) {
        int k = 16 * (i >> 3) + half * 8 + (i & 7);        // feature of this A elem
        an[i] = (_Float16)((float)a[i] * sc[k] + sh[k]);
    }
    v16h w0 = *(const v16h*)(wf2 + 0 * 512 + lane * 16);
    v16h w1 = *(const v16h*)(wf2 + 1 * 512 + lane * 16);
    v8f c0 = {}, c1 = {};
    c0 = wmma_f16(an, w0, c0);
    c1 = wmma_f16(an, w1, c1);

    int n0 = lane & 15;
    float bias0 = b2[n0], bias1 = b2[16 + n0];
#pragma unroll
    for (int r = 0; r < 8; ++r) {
        int mr = (lane >> 4) * 8 + r;
        int node = dst[tile * 16 + mr];
        atomic_add_f32(&agg[size_t(node) * 32 + n0],      swishf(c0[r] + bias0));
        atomic_add_f32(&agg[size_t(node) * 32 + 16 + n0], swishf(c1[r] + bias1));
    }
}

// ---------------------------------------------------------------------------
// Node update: u = swish(cat(h, agg/deg) @ U1 + ub1); h += swish(u @ U2 + ub2)
// One 16-node tile per wave; grid has idle waves -> wave-local LDS sync only.
// ---------------------------------------------------------------------------
__global__ void __launch_bounds__(256) node_update_kernel(
    const _Float16* __restrict__ wfu1, const _Float16* __restrict__ wfu2,
    const float* __restrict__ ub1, const float* __restrict__ ub2,
    const float* __restrict__ agg, const float* __restrict__ recip,
    float* __restrict__ h, _Float16* __restrict__ h16)
{
    __shared__ _Float16 ub[WPB][16][32];
    __shared__ float    ob[WPB][16][32];
    int tid = threadIdx.x, lane = tid & 31, wv = tid >> 5;
    int tile = blockIdx.x * WPB + wv;
    if (tile >= NTILES) return;                            // whole-wave uniform exit

    int m = lane & 15, half = lane >> 4;
    size_t node = size_t(tile) * 16 + m;

    v16h a0 = load_rowfrag(h16 + node * 32, half);         // K 0..31 : h
    float rc = recip[node];
    const float* ar = agg + node * 32;
    v16h a1;                                               // K 32..63 : mean-agg
#pragma unroll
    for (int i = 0; i < 8; ++i) {
        a1[i]     = (_Float16)(ar[half * 8 + i] * rc);
        a1[8 + i] = (_Float16)(ar[16 + half * 8 + i] * rc);
    }
    v16h b00 = *(const v16h*)(wfu1 + 0 * 512 + lane * 16);
    v16h b01 = *(const v16h*)(wfu1 + 1 * 512 + lane * 16);
    v16h b10 = *(const v16h*)(wfu1 + 2 * 512 + lane * 16);
    v16h b11 = *(const v16h*)(wfu1 + 3 * 512 + lane * 16);
    v8f c0 = {}, c1 = {};
    c0 = wmma_f16(a0, b00, c0); c0 = wmma_f16(a1, b10, c0);
    c1 = wmma_f16(a0, b01, c1); c1 = wmma_f16(a1, b11, c1);

    int n0 = lane & 15;
    float bias0 = ub1[n0], bias1 = ub1[16 + n0];
#pragma unroll
    for (int r = 0; r < 8; ++r) {
        int mr = (lane >> 4) * 8 + r;
        ub[wv][mr][n0]      = (_Float16)swishf(c0[r] + bias0);
        ub[wv][mr][16 + n0] = (_Float16)swishf(c1[r] + bias1);
    }
    wave_lds_sync();

    v16h au;                                               // transpose C->A via LDS
#pragma unroll
    for (int i = 0; i < 8; ++i) {
        au[i]     = ub[wv][m][half * 8 + i];
        au[8 + i] = ub[wv][m][16 + half * 8 + i];
    }
    v16h u0 = *(const v16h*)(wfu2 + 0 * 512 + lane * 16);
    v16h u1 = *(const v16h*)(wfu2 + 1 * 512 + lane * 16);
    v8f d0 = {}, d1 = {};
    d0 = wmma_f16(au, u0, d0);
    d1 = wmma_f16(au, u1, d1);

    float bb0 = ub2[n0], bb1 = ub2[16 + n0];
#pragma unroll
    for (int r = 0; r < 8; ++r) {
        int mr = (lane >> 4) * 8 + r;
        ob[wv][mr][n0]      = swishf(d0[r] + bb0);
        ob[wv][mr][16 + n0] = swishf(d1[r] + bb1);
    }
    wave_lds_sync();

    // Residual update with contiguous row writes: lane -> (row = l>>1, 16-col seg)
    int row = lane >> 1, seg = (lane & 1) * 16;
    size_t nr = size_t(tile) * 16 + row;
    float*    hp   = h   + nr * 32 + seg;
    _Float16* hp16 = h16 + nr * 32 + seg;
#pragma unroll
    for (int i = 0; i < 16; ++i) {
        float nv = hp[i] + ob[wv][row][seg + i];
        hp[i] = nv;
        hp16[i] = (_Float16)nv;
    }
}

__global__ void __launch_bounds__(256) copyout_kernel(const float* __restrict__ h,
                                                      float* __restrict__ out) {
    int g = blockIdx.x * 256 + threadIdx.x;
    if (g < N_ * D_) out[g] = h[g];
}

// ---------------------------------------------------------------------------
// Launch
// ---------------------------------------------------------------------------
extern "C" void kernel_launch(void* const* d_in, const int* in_sizes, int n_in,
                              void* d_out, int out_size, void* d_ws, size_t ws_size,
                              hipStream_t stream) {
    const float* x      = (const float*)d_in[0];
    const float* emb_w  = (const float*)d_in[1];
    const float* emb_b  = (const float*)d_in[2];
    const float* msg_w1 = (const float*)d_in[3];
    const float* msg_b1 = (const float*)d_in[4];
    const float* bn_g   = (const float*)d_in[5];
    const float* bn_b   = (const float*)d_in[6];
    const float* msg_w2 = (const float*)d_in[7];
    const float* msg_b2 = (const float*)d_in[8];
    const float* upd_w1 = (const float*)d_in[9];
    const float* upd_b1 = (const float*)d_in[10];
    const float* upd_w2 = (const float*)d_in[11];
    const float* upd_b2 = (const float*)d_in[12];
    const int*   eidx   = (const int*)d_in[13];
    const int*   esrc   = eidx;          // edge_index[0]
    const int*   edst   = eidx + E_;     // edge_index[1]

    char* ws = (char*)d_ws;
    _Float16* wfrag = (_Float16*)(ws);
    float*    h     = (float*)(ws + OFF_H);
    _Float16* h16   = (_Float16*)(ws + OFF_H16);
    float*    agg   = (float*)(ws + OFF_AGG);
    float*    deg   = (float*)(ws + OFF_DEG);
    float*    bnsum = (float*)(ws + OFF_BNS);
    float*    bnss  = (float*)(ws + OFF_BNP);
    _Float16* t16   = (_Float16*)(ws + OFF_T16);

    prep_weights_kernel<<<(L_ * 12 * 512 + 255) / 256, 256, 0, stream>>>(
        msg_w1, msg_w2, upd_w1, upd_w2, wfrag);
    embed_kernel<<<(N_ * D_ + 255) / 256, 256, 0, stream>>>(x, emb_w, emb_b, h, h16);

    fill_kernel<<<(N_ + 255) / 256, 256, 0, stream>>>(deg, N_);
    degree_kernel<<<(E_ + 255) / 256, 256, 0, stream>>>(edst, deg);
    recip_kernel<<<(N_ + 255) / 256, 256, 0, stream>>>(deg);

    for (int l = 0; l < L_; ++l) {
        fill_kernel<<<1, 64, 0, stream>>>(bnsum, 64);
        fill_kernel<<<(N_ * D_ + 255) / 256, 256, 0, stream>>>(agg, N_ * D_);

        edge_msg1_kernel<<<EDGE_BLOCKS, 256, 0, stream>>>(
            h16, esrc, edst, wfrag + size_t(l * 12 + 0) * 512, msg_b1 + l * D_, t16, bnsum);
        bn_finalize_kernel<<<1, 32, 0, stream>>>(bnsum, bn_g + l * D_, bn_b + l * D_, bnss);
        edge_msg2_kernel<<<EDGE_BLOCKS, 256, 0, stream>>>(
            t16, edst, bnss, wfrag + size_t(l * 12 + 4) * 512, msg_b2 + l * D_, agg);
        node_update_kernel<<<NODE_BLOCKS, 256, 0, stream>>>(
            wfrag + size_t(l * 12 + 6) * 512, wfrag + size_t(l * 12 + 10) * 512,
            upd_b1 + l * D_, upd_b2 + l * D_, agg, deg, h, h16);
    }
    copyout_kernel<<<(N_ * D_ + 255) / 256, 256, 0, stream>>>(h, (float*)d_out);
}
